// InstanceSegmentation_4131758538966
// MI455X (gfx1250) — matrix operand order repaired and checked by reading, compiled, not run on hardware
//
// MI455X (gfx1250) instance-segmentation head. All big matmuls run through
// V_WMMA_F32_16X16X32_F16 with fragments loaded DIRECTLY from global memory:
// A is [M][K] f16 row-major (lane = M-row, two 16B loads), B is pre-transposed
// [N][K] f16 row-major (lane = N-column, one 32B load). No LDS staging.
#include <hip/hip_runtime.h>
#include <hip/hip_bf16.h>

typedef _Float16 half_t;
typedef __attribute__((ext_vector_type(16))) _Float16 v16h;
typedef __attribute__((ext_vector_type(8)))  _Float16 v8h;
typedef __attribute__((ext_vector_type(8)))  float    v8f;

__device__ __forceinline__ int iclamp(int v, int lo, int hi) {
    return v < lo ? lo : (v > hi ? hi : v);
}
__device__ __forceinline__ float dsigmoid(float x) { return 1.0f / (1.0f + expf(-x)); }
__device__ __forceinline__ float dsilu(float x)    { return x / (1.0f + expf(-x)); }

// ---------------------------------------------------------------------------
// Wave-level WMMA GEMM:  C[z][m][n] = ep( sum_k A[z][m][k] * Bt[n][k] )
// A : f16 [M,K] row-major, M % 16 == 0, K % 32 == 0, rows 16B aligned.
// Bt: f16 [N,K] row-major (weights pre-transposed so K is contiguous).
// Epilogue: acc*scale[n] + bias[n] (either ptr may be null). Writes fp32 Cf
// and/or f16 Ch (row stride N). One 16x16 tile per 32-lane wave.
// ---------------------------------------------------------------------------
__global__ void k_gemm_wmma(const half_t* __restrict__ A,
                            const half_t* __restrict__ Bt,
                            float* __restrict__ Cf,
                            half_t* __restrict__ Ch,
                            const float* __restrict__ scale,
                            const float* __restrict__ bias,
                            int M, int N, int K,
                            long strideA, long strideC)
{
    const int lane = threadIdx.x;            // blockDim.x == 32
    const int hi = lane >> 4;                // K-half selector
    const int mr = lane & 15;                // row (A) / column (B) within tile
    const int n0 = blockIdx.x * 16;
    const int m0 = blockIdx.y * 16;
    const int z  = blockIdx.z;

    // per-lane contiguous source rows (A and B both use k = hi*16 + s)
    const half_t* arow = A + (size_t)z * (size_t)strideA + (size_t)(m0 + mr) * K;
    int nb = n0 + mr; if (nb > N - 1) nb = N - 1;   // clamp (stores are masked)
    const half_t* brow = Bt + (size_t)nb * K;

    v8f acc;
#pragma unroll
    for (int e = 0; e < 8; ++e) acc[e] = 0.0f;

#pragma unroll 4
    for (int k0 = 0; k0 < K; k0 += 32) {
        v8h  a0 = *(const v8h*)(arow + k0 + hi * 16);       // K = hi*16 .. +7
        v8h  a1 = *(const v8h*)(arow + k0 + hi * 16 + 8);   // K = hi*16+8 .. +15
        v16h bv = *(const v16h*)(brow + k0 + hi * 16);      // K = hi*16 .. +15
        v16h av = __builtin_shufflevector(a0, a1,
                    0, 1, 2, 3, 4, 5, 6, 7, 8, 9, 10, 11, 12, 13, 14, 15);
        acc = __builtin_amdgcn_wmma_f32_16x16x32_f16(false, av, false, bv,
                                                     (short)0, acc, false, false);
        if (k0 + 32 < K)
            __builtin_prefetch(arow + k0 + 64, 0, 1);
    }

    // D layout: lane holds column (lane&15); VGPR r holds row r + 8*(lane>=16)
    const int nCol = n0 + mr;
    if (nCol < N) {
        float sc = scale ? scale[nCol] : 1.0f;
        float bi = bias  ? bias[nCol]  : 0.0f;
#pragma unroll
        for (int r = 0; r < 8; ++r) {
            int m = m0 + r + hi * 8;
            float v = acc[r] * sc + bi;
            if (Cf) Cf[(size_t)z * (size_t)strideC + (size_t)m * N + nCol] = v;
            if (Ch) Ch[(size_t)z * (size_t)strideC + (size_t)m * N + nCol] = (half_t)v;
        }
    }
}

// LayerNorm(256) + SiLU, one row per block (blockDim == 256), f16 out
__global__ void k_ln_silu(const float* __restrict__ in, half_t* __restrict__ outH,
                          const float* __restrict__ g, const float* __restrict__ b)
{
    const int row = blockIdx.x, t = threadIdx.x;
    __shared__ float red[256];
    float x = in[(size_t)row * 256 + t];
    red[t] = x; __syncthreads();
    for (int s = 128; s > 0; s >>= 1) { if (t < s) red[t] += red[t + s]; __syncthreads(); }
    float mean = red[0] * (1.0f / 256.0f);
    __syncthreads();
    float d = x - mean;
    red[t] = d * d; __syncthreads();
    for (int s = 128; s > 0; s >>= 1) { if (t < s) red[t] += red[t + s]; __syncthreads(); }
    float var = red[0] * (1.0f / 256.0f);
    float y = g[t] * d * rsqrtf(var + 1e-5f) + b[t];
    outH[(size_t)row * 256 + t] = (half_t)dsilu(y);
}

// transpose + fp32->f16: d[z][c*R + r] = s[z][r*Cc + c]
__global__ void k_tcvt(const float* __restrict__ s, half_t* __restrict__ d,
                       int R, int Cc, long sStride, long dStride)
{
    long i = (long)blockIdx.x * 256 + threadIdx.x;
    long total = (long)R * Cc;
    if (i >= total) return;
    int z = blockIdx.z;
    int r = (int)(i / Cc), c = (int)(i % Cc);
    d[(size_t)z * dStride + (size_t)c * R + r] = (half_t)s[(size_t)z * sStride + (size_t)i];
}

__global__ void k_cvt(const float* __restrict__ s, half_t* __restrict__ d, int n)
{
    int i = blockIdx.x * 256 + threadIdx.x;
    if (i < n) d[i] = (half_t)s[i];
}

// loc head: logits[m] = h[m,:] . fw + fb   (vectorized f16 reads)
__global__ void k_gemv(const half_t* __restrict__ h, const float* __restrict__ fw,
                       const float* __restrict__ fb, float* __restrict__ out, int M)
{
    int m = blockIdx.x * 256 + threadIdx.x;
    if (m >= M) return;
    const half_t* r = h + (size_t)m * 256;
    float acc = 0.0f;
    for (int k = 0; k < 256; k += 8) {
        v8h v = *(const v8h*)(r + k);
#pragma unroll
        for (int e = 0; e < 8; ++e) acc += (float)v[e] * fw[k + e];
    }
    out[m] = acc + fb[0];
}

// per-batch iterative top-100: writes num_instances(out[b]), scores(out[4+...]), idx
__global__ void k_topk(const float* __restrict__ logits, float* __restrict__ out,
                       int* __restrict__ topidx)
{
    const int b = blockIdx.x, t = threadIdx.x;
    __shared__ float sv[5376];
    __shared__ float rv[256];
    __shared__ int   ri[256];
    for (int i = t; i < 5376; i += 256) sv[i] = logits[(size_t)b * 5376 + i];
    __syncthreads();
    int cnt = 0;
    for (int it = 0; it < 100; ++it) {
        float best = -3.0e38f; int bidx = 0x7fffffff;
        for (int i = t; i < 5376; i += 256) {
            float v = sv[i];
            if (v > best || (v == best && i < bidx)) { best = v; bidx = i; }
        }
        rv[t] = best; ri[t] = bidx;
        __syncthreads();
        for (int s = 128; s > 0; s >>= 1) {
            if (t < s) {
                float ov = rv[t + s]; int oi = ri[t + s];
                if (ov > rv[t] || (ov == rv[t] && oi < ri[t])) { rv[t] = ov; ri[t] = oi; }
            }
            __syncthreads();
        }
        float val = rv[0]; int idx = ri[0];
        if (t == 0) {
            topidx[b * 100 + it] = idx;
            out[4 + b * 100 + it] = dsigmoid(val);
            if (val > 0.0f) ++cnt;
            sv[idx] = -3.0e38f;
        }
        __syncthreads();
    }
    if (t == 0) out[b] = (float)cnt;
}

__global__ void k_gather(const half_t* __restrict__ flatH, const int* __restrict__ topidx,
                         half_t* __restrict__ g)
{
    int row = blockIdx.x;                // 0..399
    int c = threadIdx.x;                 // 0..255
    int b = row / 100;
    int idx = topidx[row];
    g[(size_t)row * 256 + c] = flatH[((size_t)b * 5376 + idx) * 256 + c];
}

__global__ void k_argmax(const float* __restrict__ logits, float* __restrict__ out, int M)
{
    int m = blockIdx.x * 256 + threadIdx.x;
    if (m >= M) return;
    const float* r = logits + (size_t)m * 80;
    float best = r[0]; int bi = 0;
    for (int c = 1; c < 80; ++c) if (r[c] > best) { best = r[c]; bi = c; }
    out[m] = (float)bi;
}

// 3x3 SAME conv (256 -> 8) + affine + SiLU; mfeat [B][4096][256] -> mf [B][4096][8]
__global__ void k_conv3x3(const float* __restrict__ mfeat, const float* __restrict__ w,
                          const float* __restrict__ s, const float* __restrict__ bb,
                          float* __restrict__ mf)
{
    int idx = blockIdx.x * 256 + threadIdx.x;     // 0..32767
    int b = blockIdx.y;
    int o = idx & 7, p = idx >> 3;
    int px = p & 63, py = p >> 6;
    float acc = 0.0f;
    for (int ky = 0; ky < 3; ++ky) {
        int yy = py + ky - 1;
        if (yy < 0 || yy > 63) continue;
        for (int kx = 0; kx < 3; ++kx) {
            int xx = px + kx - 1;
            if (xx < 0 || xx > 63) continue;
            const float* mrow = mfeat + ((size_t)b * 4096 + yy * 64 + xx) * 256;
            const float* wcol = w + (size_t)o * 256 * 9 + ky * 3 + kx;
            float a = 0.0f;
            for (int c = 0; c < 256; ++c) a += mrow[c] * wcol[(size_t)c * 9];
            acc += a;
        }
    }
    float y = acc * s[o] + bb[o];
    mf[((size_t)b * 4096 + p) * 8 + o] = dsilu(y);
}

// dynamic per-instance mask net: grid (16, 100, B), block 256
__global__ void k_dynmask(const float* __restrict__ dyn, const float* __restrict__ mf,
                          const int* __restrict__ topidx, float* __restrict__ masksLo)
{
    const int b = blockIdx.z, inst = blockIdx.y;
    const int p = blockIdx.x * 256 + threadIdx.x;          // 0..4095
    __shared__ float sp[169];
    if (threadIdx.x < 169)
        sp[threadIdx.x] = dyn[(size_t)(b * 100 + inst) * 169 + threadIdx.x];
    __syncthreads();
    int idx = topidx[b * 100 + inst];
    int w, pp;
    if (idx < 4096)      { w = 64; pp = idx; }
    else if (idx < 5120) { w = 32; pp = idx - 4096; }
    else                 { w = 16; pp = idx - 5120; }
    float offx = ((pp % w) + 0.5f) / (float)w;
    float offy = ((pp / w) + 0.5f) / (float)w;
    int px = p & 63, py = p >> 6;
    float f[10];
    const float* mfp = mf + ((size_t)b * 4096 + p) * 8;
#pragma unroll
    for (int c = 0; c < 8; ++c) f[c] = mfp[c];
    f[8] = (px + 0.5f) / 64.0f - offx;
    f[9] = (py + 0.5f) / 64.0f - offy;
    float h1[8];
#pragma unroll
    for (int d = 0; d < 8; ++d) {
        float a = sp[80 + d];
        for (int c = 0; c < 10; ++c) a += f[c] * sp[c * 8 + d];
        h1[d] = dsilu(a);
    }
    float h2[8];
#pragma unroll
    for (int d = 0; d < 8; ++d) {
        float a = sp[152 + d];
        for (int c = 0; c < 8; ++c) a += h1[c] * sp[88 + c * 8 + d];
        h2[d] = dsilu(a);
    }
    float a = sp[168];
#pragma unroll
    for (int c = 0; c < 8; ++c) a += h2[c] * sp[160 + c];
    masksLo[(size_t)(b * 100 + inst) * 4096 + p] = dsigmoid(a);
}

// bilinear 64x64 -> 512x512, align-corners-false centers, edge clamp
__global__ void k_resize(const float* __restrict__ lo, float* __restrict__ out)
{
    int tid = blockIdx.x * 256 + threadIdx.x;              // 0..262143
    int inst = blockIdx.y, b = blockIdx.z;
    int oy = tid >> 9, ox = tid & 511;
    float fy = (oy + 0.5f) * 0.125f - 0.5f;
    float fx = (ox + 0.5f) * 0.125f - 0.5f;
    int y0 = (int)floorf(fy), x0 = (int)floorf(fx);
    float wy = fy - (float)y0, wx = fx - (float)x0;
    int y0c = iclamp(y0, 0, 63), y1c = iclamp(y0 + 1, 0, 63);
    int x0c = iclamp(x0, 0, 63), x1c = iclamp(x0 + 1, 0, 63);
    const float* src = lo + (size_t)(b * 100 + inst) * 4096;
    float v00 = src[y0c * 64 + x0c], v01 = src[y0c * 64 + x1c];
    float v10 = src[y1c * 64 + x0c], v11 = src[y1c * 64 + x1c];
    float v = v00 * (1.0f - wy) * (1.0f - wx) + v01 * (1.0f - wy) * wx
            + v10 * wy * (1.0f - wx) + v11 * wy * wx;
    out[(size_t)(b * 100 + inst) * 262144 + (size_t)oy * 512 + ox] = v;
}

// ---------------------------------------------------------------------------
extern "C" void kernel_launch(void* const* d_in, const int* in_sizes, int n_in,
                              void* d_out, int out_size, void* d_ws, size_t ws_size,
                              hipStream_t stream)
{
    (void)in_sizes; (void)n_in; (void)out_size; (void)ws_size;
    const float* x3      = (const float*)d_in[3];
    const float* x4      = (const float*)d_in[4];
    const float* x5      = (const float*)d_in[5];
    const float* lat3_w  = (const float*)d_in[6];
    const float* lat3_s  = (const float*)d_in[7];
    const float* lat3_b  = (const float*)d_in[8];
    const float* lat4_w  = (const float*)d_in[9];
    const float* lat4_s  = (const float*)d_in[10];
    const float* lat4_b  = (const float*)d_in[11];
    const float* lat5_w  = (const float*)d_in[12];
    const float* lat5_s  = (const float*)d_in[13];
    const float* lat5_b  = (const float*)d_in[14];
    const float* mlat_w  = (const float*)d_in[15];
    const float* mlat_s  = (const float*)d_in[16];
    const float* mlat_b  = (const float*)d_in[17];
    const float* mhead_w = (const float*)d_in[18];
    const float* mhead_s = (const float*)d_in[19];
    const float* mhead_b = (const float*)d_in[20];
    const float* loc_hw  = (const float*)d_in[21];
    const float* loc_hb  = (const float*)d_in[22];
    const float* loc_lng = (const float*)d_in[23];
    const float* loc_lnb = (const float*)d_in[24];
    const float* loc_fw  = (const float*)d_in[25];
    const float* loc_fb  = (const float*)d_in[26];
    const float* cls_hw  = (const float*)d_in[27];
    const float* cls_hb  = (const float*)d_in[28];
    const float* cls_lng = (const float*)d_in[29];
    const float* cls_lnb = (const float*)d_in[30];
    const float* cls_fw  = (const float*)d_in[31];
    const float* cls_fb  = (const float*)d_in[32];
    const float* ker_hw  = (const float*)d_in[33];
    const float* ker_hb  = (const float*)d_in[34];
    const float* ker_lng = (const float*)d_in[35];
    const float* ker_lnb = (const float*)d_in[36];
    const float* ker_fw  = (const float*)d_in[37];
    const float* ker_fb  = (const float*)d_in[38];
    float* out = (float*)d_out;

    const int B = 4, C = 256, L3 = 4096, L4 = 1024, L5 = 256, LL = 5376;
    const int Mtok = B * LL;                         // 21504

    size_t cur = 0;
    auto carve = [&](size_t bytes) -> void* {
        void* p = (char*)d_ws + cur;
        cur += (bytes + 255) & ~(size_t)255;
        return p;
    };
    half_t* A3h    = (half_t*)carve((size_t)B * L3 * 256  * 2);
    half_t* A4h    = (half_t*)carve((size_t)B * L4 * 512  * 2);
    half_t* A5h    = (half_t*)carve((size_t)B * L5 * 1024 * 2);
    half_t* flatH  = (half_t*)carve((size_t)Mtok * C * 2);
    half_t* hA     = (half_t*)carve((size_t)Mtok * C * 2);
    half_t* hB     = (half_t*)carve((size_t)Mtok * C * 2);
    float*  tmpF   = (float*) carve((size_t)Mtok * C * 4); // also reused as mfeat
    half_t* wLat3h = (half_t*)carve((size_t)256  * 256 * 2);   // [N][K]
    half_t* wLat4h = (half_t*)carve((size_t)256  * 512 * 2);   // [N][K]
    half_t* wLat5h = (half_t*)carve((size_t)256  * 1024 * 2);  // [N][K]
    half_t* wMlath = (half_t*)carve((size_t)256  * 256 * 2);   // [N][K]
    half_t* wLocH  = (half_t*)carve((size_t)4 * 256 * 256 * 2); // [N][K] per layer
    half_t* wClsH  = (half_t*)carve((size_t)4 * 256 * 256 * 2);
    half_t* wKerH  = (half_t*)carve((size_t)4 * 256 * 256 * 2);
    half_t* wClsFh = (half_t*)carve((size_t)80  * 256 * 2);     // [N][K]
    half_t* wKerFh = (half_t*)carve((size_t)169 * 256 * 2);     // [N][K]
    float*  logitsF= (float*) carve((size_t)Mtok * 4);
    int*    topidxI= (int*)   carve((size_t)400 * 4);
    half_t* gfeatH = (half_t*)carve((size_t)400 * 256 * 2);
    half_t* g400A  = (half_t*)carve((size_t)400 * 256 * 2);
    half_t* g400B  = (half_t*)carve((size_t)400 * 256 * 2);
    float*  t400F  = (float*) carve((size_t)400 * 256 * 4);
    float*  clsLogF= (float*) carve((size_t)400 * 80  * 4);
    float*  dynF   = (float*) carve((size_t)400 * 169 * 4);
    float*  mfB    = (float*) carve((size_t)B * 4096 * 8 * 4);
    float*  masksLo= (float*) carve((size_t)400 * 4096 * 4);

    // ---- prep: activations NCHW -> [pixels, Cin] f16 (transpose) ----
    k_tcvt<<<dim3((256u * 4096u) / 256, 1, B), 256, 0, stream>>>(x3, A3h, 256, 4096,
        (long)256 * 4096, (long)4096 * 256);
    k_tcvt<<<dim3((512u * 1024u) / 256, 1, B), 256, 0, stream>>>(x4, A4h, 512, 1024,
        (long)512 * 1024, (long)1024 * 512);
    k_tcvt<<<dim3((1024u * 256u) / 256, 1, B), 256, 0, stream>>>(x5, A5h, 1024, 256,
        (long)1024 * 256, (long)256 * 1024);
    // ---- weights: lateral/mlat (Cout,Cin) are already [N][K] -> straight cvt ----
    k_cvt<<<(256 * 256) / 256, 256, 0, stream>>>(lat3_w, wLat3h, 256 * 256);
    k_cvt<<<(256 * 512) / 256, 256, 0, stream>>>(lat4_w, wLat4h, 256 * 512);
    k_cvt<<<(256 * 1024) / 256, 256, 0, stream>>>(lat5_w, wLat5h, 256 * 1024);
    k_cvt<<<(256 * 256) / 256, 256, 0, stream>>>(mlat_w, wMlath, 256 * 256);
    // ---- MLP weights (K,N) -> transpose to [N][K] ----
    k_tcvt<<<dim3((256u * 256u) / 256, 1, 4), 256, 0, stream>>>(loc_hw, wLocH, 256, 256,
        (long)65536, (long)65536);
    k_tcvt<<<dim3((256u * 256u) / 256, 1, 4), 256, 0, stream>>>(cls_hw, wClsH, 256, 256,
        (long)65536, (long)65536);
    k_tcvt<<<dim3((256u * 256u) / 256, 1, 4), 256, 0, stream>>>(ker_hw, wKerH, 256, 256,
        (long)65536, (long)65536);
    k_tcvt<<<dim3((256u * 80u) / 256, 1, 1), 256, 0, stream>>>(cls_fw, wClsFh, 256, 80, 0, 0);
    k_tcvt<<<dim3((256u * 169u) / 256, 1, 1), 256, 0, stream>>>(ker_fw, wKerFh, 256, 169, 0, 0);

    // ---- lateral 1x1 conv + BN GEMMs -> flatH [B][5376][256] (f16) ----
    k_gemm_wmma<<<dim3(16, L3 / 16, B), 32, 0, stream>>>(A3h, wLat3h, nullptr,
        flatH, lat3_s, lat3_b, L3, 256, 256, (long)L3 * 256, (long)LL * 256);
    k_gemm_wmma<<<dim3(16, L4 / 16, B), 32, 0, stream>>>(A4h, wLat4h, nullptr,
        flatH + (size_t)L3 * 256, lat4_s, lat4_b, L4, 256, 512, (long)L4 * 512, (long)LL * 256);
    k_gemm_wmma<<<dim3(16, L5 / 16, B), 32, 0, stream>>>(A5h, wLat5h, nullptr,
        flatH + (size_t)(L3 + L4) * 256, lat5_s, lat5_b, L5, 256, 1024, (long)L5 * 1024, (long)LL * 256);

    // ---- loc MLP (4 x [Linear -> LN -> SiLU]) over all tokens ----
    const half_t* locIn[5] = { flatH, hA, hB, hA, hB };
    for (int l = 0; l < 4; ++l) {
        half_t* dst = (l & 1) ? hB : hA;
        k_gemm_wmma<<<dim3(16, Mtok / 16, 1), 32, 0, stream>>>(locIn[l], wLocH + (size_t)l * 65536,
            tmpF, nullptr, nullptr, loc_hb + l * 256, Mtok, 256, 256, 0, 0);
        k_ln_silu<<<Mtok, 256, 0, stream>>>(tmpF, dst, loc_lng + l * 256, loc_lnb + l * 256);
    }
    k_gemv<<<(Mtok + 255) / 256, 256, 0, stream>>>(hB, loc_fw, loc_fb, logitsF, Mtok);

    // ---- top-100 per batch: num_instances + scores + indices ----
    k_topk<<<B, 256, 0, stream>>>(logitsF, out, topidxI);
    k_gather<<<400, 256, 0, stream>>>(flatH, topidxI, gfeatH);

    // ---- cls MLP over 400 tokens ----
    const half_t* clsIn[5] = { gfeatH, g400A, g400B, g400A, g400B };
    for (int l = 0; l < 4; ++l) {
        half_t* dst = (l & 1) ? g400B : g400A;
        k_gemm_wmma<<<dim3(16, 25, 1), 32, 0, stream>>>(clsIn[l], wClsH + (size_t)l * 65536,
            t400F, nullptr, nullptr, cls_hb + l * 256, 400, 256, 256, 0, 0);
        k_ln_silu<<<400, 256, 0, stream>>>(t400F, dst, cls_lng + l * 256, cls_lnb + l * 256);
    }
    k_gemm_wmma<<<dim3(5, 25, 1), 32, 0, stream>>>(g400B, wClsFh, clsLogF, nullptr,
        nullptr, cls_fb, 400, 80, 256, 0, 0);
    k_argmax<<<(400 + 255) / 256, 256, 0, stream>>>(clsLogF, out + 404, 400);

    // ---- kernel-param MLP over 400 tokens -> dyn [400][169] ----
    const half_t* kerIn[5] = { gfeatH, g400A, g400B, g400A, g400B };
    for (int l = 0; l < 4; ++l) {
        half_t* dst = (l & 1) ? g400B : g400A;
        k_gemm_wmma<<<dim3(16, 25, 1), 32, 0, stream>>>(kerIn[l], wKerH + (size_t)l * 65536,
            t400F, nullptr, nullptr, ker_hb + l * 256, 400, 256, 256, 0, 0);
        k_ln_silu<<<400, 256, 0, stream>>>(t400F, dst, ker_lng + l * 256, ker_lnb + l * 256);
    }
    k_gemm_wmma<<<dim3(11, 25, 1), 32, 0, stream>>>(g400B, wKerFh, dynF, nullptr,
        nullptr, ker_fb, 400, 169, 256, 0, 0);

    // ---- mask branch: mlat GEMM (reuse tmpF as mfeat [B][4096][256]) + 3x3 conv ----
    k_gemm_wmma<<<dim3(16, L3 / 16, B), 32, 0, stream>>>(A3h, wMlath, tmpF, nullptr,
        mlat_s, mlat_b, L3, 256, 256, (long)L3 * 256, (long)L3 * 256);
    k_conv3x3<<<dim3(128, B), 256, 0, stream>>>(tmpF, mhead_w, mhead_s, mhead_b, mfB);

    // ---- dynamic per-instance masks + upsample to 512x512 ----
    k_dynmask<<<dim3(16, 100, B), 256, 0, stream>>>(dynF, mfB, topidxI, masksLo);
    k_resize<<<dim3(1024, 100, B), 256, 0, stream>>>(masksLo, out + 804);
}
